// ReadmissionEncoder_58239756533864
// MI455X (gfx1250) — compile-verified, matrix operand order checked by reading, not run on hardware
//
#include <hip/hip_runtime.h>
#include <hip/hip_bf16.h>
#include <math.h>

// ---- problem constants ----
#define D_MODEL 768
#define NHEAD   12
#define DH      64
#define NLAYERS 12
#define DFF     3072
#define BATCH   256
#define TT      30
#define DFEAT   256
#define LTOK    60          // T*M
#define SEQ     62          // L + 2 special tokens
#define SPAD    64          // padded seq for WMMA

typedef float  v8f   __attribute__((ext_vector_type(8)));
typedef __bf16 v16bf __attribute__((ext_vector_type(16)));
typedef __bf16 v8bf  __attribute__((ext_vector_type(8)));
typedef unsigned int u32x4 __attribute__((ext_vector_type(4)));
typedef int          i32x4 __attribute__((ext_vector_type(4)));
typedef int          i32x8 __attribute__((ext_vector_type(8)));

__device__ inline __bf16 to_bf16(float f) {
  unsigned u = __builtin_bit_cast(unsigned, f);
  unsigned r = u + 0x7FFFu + ((u >> 16) & 1u);      // round-to-nearest-even
  return __builtin_bit_cast(__bf16, (unsigned short)(r >> 16));
}

__device__ inline float gelu_exact(float x) {
  return 0.5f * x * (1.0f + erff(x * 0.70710678118654752440f));
}

// Load one 16x32 bf16 WMMA fragment from LDS (row-major, stride_elems bf16 per row).
__device__ inline v16bf ld_frag(const __bf16* base, int stride_elems, int lane) {
  int r  = lane & 15;
  int ks = (lane >> 4) << 3;                         // 0 or 8
  const __bf16* p = base + r * stride_elems + ks;
  v8bf lo = *(const v8bf*)p;
  v8bf hi = *(const v8bf*)(p + 16);
  return __builtin_shufflevector(lo, hi, 0,1,2,3,4,5,6,7,8,9,10,11,12,13,14,15);
}

// ---- Tensor Data Mover: async 2D bf16 tile (width x rows, row stride = stride_elems) ----
// LDS destination gets 16 pad bytes after every 64 data bytes (pad_interval=16 DW, pad_amount=4 DW)
// -> LDS row pitch = 80 B = 40 bf16, matching AST/WST below. Tracked by TENSORcnt.
__device__ inline void tdm_load_2d_bf16(unsigned lds_addr, const __bf16* gptr,
                                        int width_elems, int rows, int stride_elems) {
  unsigned long long ga = (unsigned long long)(size_t)gptr;
  u32x4 g0;
  g0[0] = 1u;                                            // count=1, user mode, no gather
  g0[1] = lds_addr;                                      // lds_addr [63:32]
  g0[2] = (unsigned)(ga & 0xFFFFFFFFu);                  // global_addr lo
  g0[3] = (unsigned)((ga >> 32) & 0x01FFFFFFu) | (2u << 30); // addr hi | type=2
  i32x8 g1;
  g1[0] = (int)((1u << 16)      // data_size = 2 bytes
              | (1u << 20)      // pad_enable
              | (3u << 22)      // pad_interval: 16 DWORDs
              | (3u << 25));    // pad_amount : 4 DWORDs
  g1[1] = (int)((unsigned)width_elems << 16);            // tensor_dim0 lo16 @ [63:48]
  g1[2] = (int)((unsigned)rows << 16);                   // tensor_dim1 lo16 @ [95:80]
  g1[3] = (int)((unsigned)width_elems << 16);            // tile_dim0 @ [127:112]
  g1[4] = (int)(unsigned)rows;                           // tile_dim1 @ [143:128]
  g1[5] = stride_elems;                                  // tensor_dim0_stride lo32
  g1[6] = 0;
  g1[7] = 0;
  i32x4 z4 = {0, 0, 0, 0};
#if defined(__clang_major__) && (__clang_major__ >= 23)
  i32x8 z8 = {0, 0, 0, 0, 0, 0, 0, 0};
  __builtin_amdgcn_tensor_load_to_lds(g0, g1, z4, z4, z8, 0);
#else
  __builtin_amdgcn_tensor_load_to_lds(g0, g1, z4, z4, 0);
#endif
}

// ---------------- f32 -> bf16 bulk convert ----------------
__global__ __launch_bounds__(256) void cvt_bf16_kernel(const float* __restrict__ src,
                                                       __bf16* __restrict__ dst, size_t n) {
  size_t i = (size_t)blockIdx.x * 256 + threadIdx.x;
  size_t stride = (size_t)gridDim.x * 256;
  for (; i < n; i += stride) dst[i] = to_bf16(src[i]);
}

// ---------------- WMMA GEMM, double-buffered TDM tile DMA ----------------
// C[M,N] = act( A[M,K](bf16) @ W[N,K](bf16)^T + bias (+resid) ), C = f32 or bf16
// Block tile 128x128, 8 waves, each wave 32x64 (2x4 WMMA tiles), BK=32.
#define BM 128
#define BN 128
#define BK 32
#define AST 40    // LDS row stride (bf16): 32 data + 8 pad = 80B rows (TDM pad)

template<int ACT, bool RESID, bool OUTBF>
__global__ __launch_bounds__(256) void gemm_tdm(
    const __bf16* __restrict__ A, const __bf16* __restrict__ W,
    const float* __restrict__ bias, const float* __restrict__ resid,
    void* __restrict__ Cv, int M, int N, int K)
{
  __shared__ __align__(16) __bf16 As[2][BM * AST];
  __shared__ __align__(16) __bf16 Ws[2][BN * AST];
  int tid  = threadIdx.x;
  int lane = tid & 31;
  int wave = tid >> 5;                 // 8 waves
  int wm   = wave & 3;                 // 4 along M (32 rows each)
  int wn   = wave >> 2;                // 2 along N (64 cols each)
  int gm0  = blockIdx.y * BM;
  int gn0  = blockIdx.x * BN;
  unsigned lds_a[2] = { (unsigned)(size_t)&As[0][0], (unsigned)(size_t)&As[1][0] };
  unsigned lds_w[2] = { (unsigned)(size_t)&Ws[0][0], (unsigned)(size_t)&Ws[1][0] };

  v8f acc[2][4] = {};

  // prologue: DMA first k-chunk into buffer 0
  if (tid < 32) {
    tdm_load_2d_bf16(lds_a[0], A + (size_t)gm0 * K, BK, BM, K);
    tdm_load_2d_bf16(lds_w[0], W + (size_t)gn0 * K, BK, BN, K);
  }

  int cur = 0;
  for (int k0 = 0; k0 < K; k0 += BK) {
    bool has_next = (k0 + BK) < K;
    if (tid < 32) {                    // wave 0 drives the Tensor Data Mover
      if (has_next) {
        tdm_load_2d_bf16(lds_a[cur ^ 1], A + (size_t)gm0 * K + (k0 + BK), BK, BM, K);
        tdm_load_2d_bf16(lds_w[cur ^ 1], W + (size_t)gn0 * K + (k0 + BK), BK, BN, K);
        __builtin_amdgcn_s_wait_tensorcnt(2);   // oldest 2 (current buffer) complete
      } else {
        __builtin_amdgcn_s_wait_tensorcnt(0);
      }
    }
    __syncthreads();

    const __bf16* ab = &As[cur][0];
    const __bf16* wb = &Ws[cur][0];
    v16bf af[2], bf[4];
    #pragma unroll
    for (int i = 0; i < 2; ++i) af[i] = ld_frag(ab + (wm * 32 + i * 16) * AST, AST, lane);
    #pragma unroll
    for (int j = 0; j < 4; ++j) bf[j] = ld_frag(wb + (wn * 64 + j * 16) * AST, AST, lane);
    #pragma unroll
    for (int i = 0; i < 2; ++i)
      #pragma unroll
      for (int j = 0; j < 4; ++j)
        acc[i][j] = __builtin_amdgcn_wmma_f32_16x16x32_bf16(
            false, af[i], false, bf[j], (short)0, acc[i][j], false, false);
    __syncthreads();                   // reads done before this buffer is overwritten
    cur ^= 1;
  }

  float*  Cf = (float*)Cv;
  __bf16* Cb = (__bf16*)Cv;
  int rofs = (lane >> 4) << 3;
  int coll = lane & 15;
  #pragma unroll
  for (int i = 0; i < 2; ++i)
    #pragma unroll
    for (int j = 0; j < 4; ++j) {
      int gn = gn0 + wn * 64 + j * 16 + coll;
      float bv = bias[gn];
      #pragma unroll
      for (int r = 0; r < 8; ++r) {
        int gm = gm0 + wm * 32 + i * 16 + r + rofs;
        float v = acc[i][j][r] + bv;
        if (RESID) v += resid[(size_t)gm * N + gn];
        if (ACT)   v = gelu_exact(v);
        size_t idx = (size_t)gm * N + gn;
        if (OUTBF) Cb[idx] = to_bf16(v); else Cf[idx] = v;
      }
    }
}

// ---------------- row LayerNorm (f32 in, bf16 out) ----------------
__global__ __launch_bounds__(256) void layernorm_bf16(
    const float* __restrict__ X, const float* __restrict__ w, const float* __restrict__ b,
    __bf16* __restrict__ Z, int ncols)
{
  int row = blockIdx.x, t = threadIdx.x;
  __shared__ float red[256];
  const float* x = X + (size_t)row * ncols;
  float s = 0.f;
  for (int c = t; c < ncols; c += 256) s += x[c];
  red[t] = s; __syncthreads();
  for (int o = 128; o > 0; o >>= 1) { if (t < o) red[t] += red[t + o]; __syncthreads(); }
  float mean = red[0] / ncols; __syncthreads();
  float v = 0.f;
  for (int c = t; c < ncols; c += 256) { float d = x[c] - mean; v += d * d; }
  red[t] = v; __syncthreads();
  for (int o = 128; o > 0; o >>= 1) { if (t < o) red[t] += red[t + o]; __syncthreads(); }
  float inv = rsqrtf(red[0] / ncols + 1e-5f);
  for (int c = t; c < ncols; c += 256)
    Z[(size_t)row * ncols + c] = to_bf16((x[c] - mean) * inv * w[c] + b[c]);
}

// ---------------- fused attention per (batch, head), WMMA both matmuls, bf16 out ----------------
__global__ __launch_bounds__(256) void attention_kernel(
    const float* __restrict__ qkv, const int* __restrict__ vlen,
    __bf16* __restrict__ obuf)
{
  __shared__ __align__(16) __bf16 Qs[SPAD * 72];   // reused for P after softmax
  __shared__ __align__(16) __bf16 Ks[SPAD * 72];
  __shared__ __align__(16) __bf16 Vt[SPAD * 72];   // V transposed: [d][s]
  __shared__ float Sc[SPAD * 64];

  int b = blockIdx.x / NHEAD;
  int hh = blockIdx.x % NHEAD;
  int tid = threadIdx.x, lane = tid & 31, wave = tid >> 5;
  int vlim = vlen[b] + 1;                           // keys with s > vlim masked

  const float* base = qkv + (size_t)b * SEQ * (3 * D_MODEL) + hh * DH;
  #pragma unroll
  for (int j = 0; j < 16; ++j) {                    // 64x64 / 256 threads
    int lin = j * 256 + tid;
    int s = lin >> 6, d = lin & 63;
    float q = 0.f, k = 0.f, vv = 0.f;
    if (s < SEQ) {
      const float* rp = base + (size_t)s * (3 * D_MODEL);
      q = rp[d]; k = rp[D_MODEL + d]; vv = rp[2 * D_MODEL + d];
    }
    Qs[s * 72 + d] = to_bf16(q);
    Ks[s * 72 + d] = to_bf16(k);
    Vt[d * 72 + s] = to_bf16(vv);
  }
  __syncthreads();

  // scores = Q @ K^T * 1/sqrt(dh) + mask
  #pragma unroll
  for (int tt = 0; tt < 2; ++tt) {
    int tile = wave * 2 + tt;
    int tm = tile >> 2, tn = tile & 3;
    v8f acc = {};
    #pragma unroll
    for (int k0 = 0; k0 < 64; k0 += 32) {
      v16bf af = ld_frag(&Qs[(tm * 16) * 72 + k0], 72, lane);
      v16bf bfr = ld_frag(&Ks[(tn * 16) * 72 + k0], 72, lane);
      acc = __builtin_amdgcn_wmma_f32_16x16x32_bf16(false, af, false, bfr, (short)0, acc, false, false);
    }
    int rofs = (lane >> 4) << 3, coll = lane & 15;
    int col = tn * 16 + coll;
    float biasv = (col > vlim) ? -1e9f : 0.f;
    #pragma unroll
    for (int r = 0; r < 8; ++r)
      Sc[(tm * 16 + r + rofs) * 64 + col] = acc[r] * 0.125f + biasv;
  }
  __syncthreads();

  // softmax: wave w owns rows 8w..8w+7 ; lane covers cols {lane, lane+32}
  #pragma unroll
  for (int r8 = 0; r8 < 8; ++r8) {
    int row = wave * 8 + r8;
    float x0 = Sc[row * 64 + lane], x1 = Sc[row * 64 + 32 + lane];
    float m = fmaxf(x0, x1);
    for (int o = 16; o > 0; o >>= 1) m = fmaxf(m, __shfl_xor(m, o, 32));
    float e0 = __expf(x0 - m), e1 = __expf(x1 - m);
    float sum = e0 + e1;
    for (int o = 16; o > 0; o >>= 1) sum += __shfl_xor(sum, o, 32);
    float inv = 1.f / sum;
    Qs[row * 72 + lane]      = to_bf16(e0 * inv);
    Qs[row * 72 + 32 + lane] = to_bf16(e1 * inv);
  }
  __syncthreads();

  // O = P @ V  (B tile = V^T in [d][s] form), bf16 out for next GEMM
  #pragma unroll
  for (int tt = 0; tt < 2; ++tt) {
    int tile = wave * 2 + tt;
    int tm = tile >> 2, tn = tile & 3;
    v8f acc = {};
    #pragma unroll
    for (int k0 = 0; k0 < 64; k0 += 32) {
      v16bf af = ld_frag(&Qs[(tm * 16) * 72 + k0], 72, lane);
      v16bf bfr = ld_frag(&Vt[(tn * 16) * 72 + k0], 72, lane);
      acc = __builtin_amdgcn_wmma_f32_16x16x32_bf16(false, af, false, bfr, (short)0, acc, false, false);
    }
    int rofs = (lane >> 4) << 3, coll = lane & 15;
    int col = tn * 16 + coll;
    #pragma unroll
    for (int r = 0; r < 8; ++r) {
      int row = tm * 16 + r + rofs;
      if (row < SEQ)
        obuf[((size_t)(b * SEQ + row)) * D_MODEL + hh * DH + col] = to_bf16(acc[r]);
    }
  }
}

// ---------------- pack valid tokens + embeddings + special tokens w/ sinusoidal PE --------------
__global__ __launch_bounds__(256) void pack_kernel(
    const float* __restrict__ emb, const int* __restrict__ mask,
    const float* __restrict__ mod_emb, const float* __restrict__ day_emb,
    const float* __restrict__ readmit, const float* __restrict__ contrast,
    float* __restrict__ h, int* __restrict__ vlen_out)
{
  int b = blockIdx.x, t = threadIdx.x;
  __shared__ int dest[LTOK];
  __shared__ int sv;
  if (t == 0) {
    int c = 0;
    for (int l = 0; l < LTOK; ++l) {
      bool valid = mask[b * TT + (l >> 1)] != 0;
      dest[l] = valid ? c++ : -1;
    }
    sv = c;
    vlen_out[b] = c;
  }
  __syncthreads();
  int vl = sv;
  float* hb = h + (size_t)b * SEQ * D_MODEL;
  for (int row = vl + 2; row < SEQ; ++row)
    for (int c = t; c < D_MODEL; c += 256) hb[row * D_MODEL + c] = 0.f;
  for (int l = 0; l < LTOK; ++l) {
    int d = dest[l];
    if (d < 0) continue;
    const float* e = emb + ((size_t)b * LTOK + l) * D_MODEL;
    for (int c = t; c < D_MODEL; c += 256)
      hb[d * D_MODEL + c] = e[c] + mod_emb[(l & 1) * D_MODEL + c] + day_emb[(l >> 1) * D_MODEL + c];
  }
  for (int c = t; c < D_MODEL; c += 256) {
    int i2 = (c >> 1) * 2;
    float freq = __expf(-(float)i2 * (9.210340371976184f / (float)D_MODEL));
    float a0 = (float)vl * freq, a1 = (float)(vl + 1) * freq;
    float p0 = (c & 1) ? __cosf(a0) : __sinf(a0);
    float p1 = (c & 1) ? __cosf(a1) : __sinf(a1);
    hb[vl * D_MODEL + c]       = readmit[c]  + p0;
    hb[(vl + 1) * D_MODEL + c] = contrast[c] + p1;
  }
}

// ---------------- readmit-token gather (bf16 out for head GEMM) ----------------
__global__ void gather_tok(const float* __restrict__ h, const int* __restrict__ vlen,
                           __bf16* __restrict__ tok)
{
  int b = blockIdx.x;
  int vl = vlen[b];
  for (int c = threadIdx.x; c < D_MODEL; c += blockDim.x)
    tok[(size_t)b * D_MODEL + c] = to_bf16(h[((size_t)b * SEQ + vl) * D_MODEL + c]);
}

// ---------------- head: LN -> gelu -> dot(hW2) + hb2 ----------------
__global__ __launch_bounds__(128) void head_final(
    const float* __restrict__ hm_in, const float* __restrict__ lnw, const float* __restrict__ lnb,
    const float* __restrict__ hW2, const float* __restrict__ hb2, float* __restrict__ out)
{
  const int NC = D_MODEL / 2;                       // 384
  int b = blockIdx.x, t = threadIdx.x;
  __shared__ float red[128];
  const float* x = hm_in + (size_t)b * NC;
  float s = 0.f;
  for (int c = t; c < NC; c += 128) s += x[c];
  red[t] = s; __syncthreads();
  for (int o = 64; o > 0; o >>= 1) { if (t < o) red[t] += red[t + o]; __syncthreads(); }
  float mean = red[0] / NC; __syncthreads();
  float v = 0.f;
  for (int c = t; c < NC; c += 128) { float d = x[c] - mean; v += d * d; }
  red[t] = v; __syncthreads();
  for (int o = 64; o > 0; o >>= 1) { if (t < o) red[t] += red[t + o]; __syncthreads(); }
  float inv = rsqrtf(red[0] / NC + 1e-5f); __syncthreads();
  float dot = 0.f;
  for (int c = t; c < NC; c += 128) {
    float z = (x[c] - mean) * inv * lnw[c] + lnb[c];
    dot += gelu_exact(z) * hW2[c];
  }
  red[t] = dot; __syncthreads();
  for (int o = 64; o > 0; o >>= 1) { if (t < o) red[t] += red[t + o]; __syncthreads(); }
  if (t == 0) out[b] = red[0] + hb2[0];
}

// ---------------- host driver ----------------
extern "C" void kernel_launch(void* const* d_in, const int* in_sizes, int n_in,
                              void* d_out, int out_size, void* d_ws, size_t ws_size,
                              hipStream_t stream)
{
  (void)in_sizes; (void)n_in; (void)out_size; (void)ws_size;
  const float* x        = (const float*)d_in[0];
  const int*   mask     = (const int*)d_in[1];
  const float* Wp       = (const float*)d_in[2];
  const float* bp       = (const float*)d_in[3];
  const float* mod_emb  = (const float*)d_in[4];
  const float* day_emb  = (const float*)d_in[5];
  const float* readmit  = (const float*)d_in[6];
  const float* contrast = (const float*)d_in[7];
  const float* WqkvS    = (const float*)d_in[8];
  const float* bqkvS    = (const float*)d_in[9];
  const float* WoS      = (const float*)d_in[10];
  const float* boS      = (const float*)d_in[11];
  const float* ln1wS    = (const float*)d_in[12];
  const float* ln1bS    = (const float*)d_in[13];
  const float* ln2wS    = (const float*)d_in[14];
  const float* ln2bS    = (const float*)d_in[15];
  const float* W1S      = (const float*)d_in[16];
  const float* b1S      = (const float*)d_in[17];
  const float* W2S      = (const float*)d_in[18];
  const float* b2S      = (const float*)d_in[19];
  const float* hW1      = (const float*)d_in[20];
  const float* hb1      = (const float*)d_in[21];
  const float* hlnw     = (const float*)d_in[22];
  const float* hlnb     = (const float*)d_in[23];
  const float* hW2      = (const float*)d_in[24];
  const float* hb2      = (const float*)d_in[25];
  float* out = (float*)d_out;

  const int Mtok = BATCH * SEQ;                     // 15872 (div by 128)
  char* ws = (char*)d_ws;
  size_t off = 0;
  auto allocF = [&](size_t n) { float* p = (float*)(ws + off); off += n * sizeof(float); return p; };
  auto allocB = [&](size_t n) { __bf16* p = (__bf16*)(ws + off); off += ((n * sizeof(__bf16) + 15) & ~(size_t)15); return p; };

  // f32 buffers
  float* h    = allocF((size_t)Mtok * D_MODEL);
  float* qkv  = allocF((size_t)Mtok * 3 * D_MODEL);
  float* hm   = allocF((size_t)BATCH * (D_MODEL / 2));
  int*   vlen = (int*)(ws + off); off += ((BATCH * sizeof(int) + 15) & ~(size_t)15);
  float* emb  = qkv;   // alias: emb (15360x768 f32) consumed by pack before qkv first written
  // bf16 operand buffers
  __bf16* xbf   = allocB((size_t)BATCH * LTOK * DFEAT);
  __bf16* z     = allocB((size_t)Mtok * D_MODEL);
  __bf16* obuf  = allocB((size_t)Mtok * D_MODEL);
  __bf16* ffb   = allocB((size_t)Mtok * DFF);
  __bf16* tok   = allocB((size_t)BATCH * D_MODEL);
  // bf16 weights (converted once per launch)
  __bf16* Wp_b   = allocB((size_t)D_MODEL * DFEAT);
  __bf16* Wqkv_b = allocB((size_t)NLAYERS * 3 * D_MODEL * D_MODEL);
  __bf16* Wo_b   = allocB((size_t)NLAYERS * D_MODEL * D_MODEL);
  __bf16* W1_b   = allocB((size_t)NLAYERS * DFF * D_MODEL);
  __bf16* W2_b   = allocB((size_t)NLAYERS * D_MODEL * DFF);
  __bf16* hW1_b  = allocB((size_t)(D_MODEL / 2) * D_MODEL);

  dim3 blk(256);
  auto cvt = [&](const float* s, __bf16* d, size_t n) {
    int g = (int)((n + 256 * 8 - 1) / (256 * 8)); if (g > 4096) g = 4096;
    cvt_bf16_kernel<<<g, blk, 0, stream>>>(s, d, n);
  };
  cvt(x,     xbf,    (size_t)BATCH * LTOK * DFEAT);
  cvt(Wp,    Wp_b,   (size_t)D_MODEL * DFEAT);
  cvt(WqkvS, Wqkv_b, (size_t)NLAYERS * 3 * D_MODEL * D_MODEL);
  cvt(WoS,   Wo_b,   (size_t)NLAYERS * D_MODEL * D_MODEL);
  cvt(W1S,   W1_b,   (size_t)NLAYERS * DFF * D_MODEL);
  cvt(W2S,   W2_b,   (size_t)NLAYERS * D_MODEL * DFF);
  cvt(hW1,   hW1_b,  (size_t)(D_MODEL / 2) * D_MODEL);

  // 1) token projection: gelu(x @ Wp^T + bp) -> emb [15360 x 768] f32
  gemm_tdm<1, false, false><<<dim3(D_MODEL / BN, (BATCH * LTOK) / BM), blk, 0, stream>>>(
      xbf, Wp_b, bp, nullptr, emb, BATCH * LTOK, D_MODEL, DFEAT);
  // 2) ragged pack + embeddings + special tokens -> h f32
  pack_kernel<<<BATCH, blk, 0, stream>>>(emb, mask, mod_emb, day_emb, readmit, contrast, h, vlen);

  for (int l = 0; l < NLAYERS; ++l) {
    const __bf16* Wqkv = Wqkv_b + (size_t)l * 3 * D_MODEL * D_MODEL;
    const __bf16* Wo   = Wo_b   + (size_t)l * D_MODEL * D_MODEL;
    const __bf16* W1   = W1_b   + (size_t)l * DFF * D_MODEL;
    const __bf16* W2   = W2_b   + (size_t)l * D_MODEL * DFF;
    const float*  bqkv = bqkvS  + (size_t)l * 3 * D_MODEL;
    const float*  bo   = boS    + (size_t)l * D_MODEL;
    const float*  b1   = b1S    + (size_t)l * DFF;
    const float*  b2   = b2S    + (size_t)l * D_MODEL;

    layernorm_bf16<<<Mtok, blk, 0, stream>>>(h, ln1wS + l * D_MODEL, ln1bS + l * D_MODEL, z, D_MODEL);
    gemm_tdm<0, false, false><<<dim3(3 * D_MODEL / BN, Mtok / BM), blk, 0, stream>>>(
        z, Wqkv, bqkv, nullptr, qkv, Mtok, 3 * D_MODEL, D_MODEL);
    attention_kernel<<<BATCH * NHEAD, blk, 0, stream>>>(qkv, vlen, obuf);
    gemm_tdm<0, true, false><<<dim3(D_MODEL / BN, Mtok / BM), blk, 0, stream>>>(
        obuf, Wo, bo, h, h, Mtok, D_MODEL, D_MODEL);
    layernorm_bf16<<<Mtok, blk, 0, stream>>>(h, ln2wS + l * D_MODEL, ln2bS + l * D_MODEL, z, D_MODEL);
    gemm_tdm<1, false, true><<<dim3(DFF / BN, Mtok / BM), blk, 0, stream>>>(
        z, W1, b1, nullptr, ffb, Mtok, DFF, D_MODEL);
    gemm_tdm<0, true, false><<<dim3(D_MODEL / BN, Mtok / BM), blk, 0, stream>>>(
        ffb, W2, b2, h, h, Mtok, D_MODEL, DFF);
  }

  gather_tok<<<BATCH, blk, 0, stream>>>(h, vlen, tok);
  gemm_tdm<0, false, false><<<dim3((D_MODEL / 2) / BN, BATCH / BM), blk, 0, stream>>>(
      tok, hW1_b, hb1, nullptr, hm, BATCH, D_MODEL / 2, D_MODEL);
  head_final<<<BATCH, dim3(128), 0, stream>>>(hm, hlnw, hlnb, hW2, hb2, out);
}
